// MultiHeadAttentionClassical_65481071395783
// MI455X (gfx1250) — compile-verified, matrix operand order checked by reading, not run on hardware
//
#include <hip/hip_runtime.h>

// ---------------------------------------------------------------------------
// MultiHeadAttention (B=4, S=2048, E=1024, H=16, D=64), fp32 in/out.
// bf16 WMMA (v_wmma_f32_16x16x32_bf16) with f32 accumulation everywhere.
// Attention computes scores transposed (S^T = K * Q^T) so the probability
// C-fragments are bit-identical to the B-fragment layout needed for
// O^T = V^T * P^T: no transposes, no cross-lane movement.
// K/V tiles are staged into LDS once per 32-key step by the whole workgroup
// using gfx1250 async LDS loads (global_load_async_to_lds_b128 + ASYNCcnt),
// giving 8x reuse across the 8 waves of the block.
// ---------------------------------------------------------------------------

typedef __bf16 bf16_t;
typedef __attribute__((ext_vector_type(8)))  __bf16 v8bf;
typedef __attribute__((ext_vector_type(16))) __bf16 v16bf;
typedef __attribute__((ext_vector_type(8)))  float  v8f;

#define EMBED 1024
#define HEADS 16
#define HDIM  64
#define BATCH 4
#define SEQ   2048
#define MROWS (BATCH * SEQ)     // 8192

__device__ __forceinline__ v8f vzero8f() {
  v8f z;
#pragma unroll
  for (int i = 0; i < 8; ++i) z[i] = 0.0f;
  return z;
}

// 16-bit 16x32 A/B fragment load from a row-major matrix (ld in elements).
//   lanes 0-15  : row0+lane,    K = {k0..k0+7} then {k0+16..k0+23}
//   lanes 16-31 : row0+lane-16, K = {k0+8..k0+15} then {k0+24..k0+31}
// Two 16-byte chunks per lane.  Works for global or LDS pointers.
__device__ __forceinline__ v16bf load_frag16(const bf16_t* base,
                                             int ld, int row0, int k0) {
  const int lane = threadIdx.x & 31;
  const int half = lane >> 4;
  const int l = lane & 15;
  const bf16_t* p = base + (size_t)(row0 + l) * (size_t)ld + k0 + half * 8;
  v8bf lo = *(const v8bf*)p;
  v8bf hi = *(const v8bf*)(p + 16);
  v16bf r;
#pragma unroll
  for (int i = 0; i < 8; ++i) { r[i] = lo[i]; r[8 + i] = hi[i]; }
  return r;
}

__device__ __forceinline__ v8f wmma_bf16(v16bf a, v16bf b, v8f c) {
  return __builtin_amdgcn_wmma_f32_16x16x32_bf16(false, a, false, b,
                                                 (short)0, c, false, false);
}

// gfx1250 async global->LDS copy, 16 bytes per call.  lds_off is the LDS byte
// address (low 32 bits of the generic pointer), gptr the 64-bit global address.
__device__ __forceinline__ void async_copy_b128(const bf16_t* lds_ptr,
                                                const bf16_t* gptr) {
  unsigned lds_off = (unsigned)(uintptr_t)lds_ptr;
  asm volatile("global_load_async_to_lds_b128 %0, %1, off"
               :: "v"(lds_off), "v"(gptr) : "memory");
}
__device__ __forceinline__ void wait_asynccnt0() {
  asm volatile("s_wait_asynccnt 0x0" ::: "memory");
}

// ---------------------------------------------------------------------------
// fp32 -> bf16 conversion
// ---------------------------------------------------------------------------
__global__ void cvt_f32_bf16_kernel(const float* __restrict__ in,
                                    bf16_t* __restrict__ out, int n) {
  int i = blockIdx.x * blockDim.x + threadIdx.x;
  if (i < n) out[i] = (bf16_t)in[i];
}

// key-padding mask -> additive bias (0 or -1e9)
__global__ void mask_bias_kernel(const int* __restrict__ mask,
                                 float* __restrict__ madd, int n) {
  int i = blockIdx.x * blockDim.x + threadIdx.x;
  if (i < n) madd[i] = (mask[i] != 0) ? 0.0f : -1e9f;
}

// ---------------------------------------------------------------------------
// QKV projection: qkv[m, f] = sum_e x[m, e] * w_qkv[f, e]
// One wave computes a 32(M) x 64(N) tile: each weight B-fragment feeds two
// WMMAs (halves weight-side bandwidth per FLOP).  Scatter:
//   f in [0,1024)    -> Q[b,h,s,d]    (row-major, bf16)
//   f in [1024,2048) -> K[b,h,s,d]    (row-major, bf16)
//   f in [2048,3072) -> Vt[b,h,d,s]   (transposed, bf16)
// ---------------------------------------------------------------------------
__global__ __launch_bounds__(256) void qkv_gemm_kernel(
    const bf16_t* __restrict__ xbf, const bf16_t* __restrict__ wbf,
    bf16_t* __restrict__ Qbf, bf16_t* __restrict__ Kbf,
    bf16_t* __restrict__ Vt) {
  const int wid = blockIdx.x * 8 + (threadIdx.x >> 5);
  const int lane = threadIdx.x & 31;
  const int mt = wid & 255;          // 256 M tiles of 32 rows
  const int nt = wid >> 8;           // 48 N tiles (3072/64)
  const int m0 = mt * 32, n0 = nt * 64;

  v8f c0[4], c1[4];
#pragma unroll
  for (int t = 0; t < 4; ++t) { c0[t] = vzero8f(); c1[t] = vzero8f(); }

  for (int k = 0; k < EMBED; k += 32) {
    v16bf a0 = load_frag16(xbf, EMBED, m0, k);
    v16bf a1 = load_frag16(xbf, EMBED, m0 + 16, k);
#pragma unroll
    for (int t = 0; t < 4; ++t) {
      v16bf bb = load_frag16(wbf, EMBED, n0 + 16 * t, k);
      c0[t] = wmma_bf16(a0, bb, c0[t]);
      c1[t] = wmma_bf16(a1, bb, c1[t]);
    }
  }

  const int sec = n0 >> 10;          // 0=Q 1=K 2=V
  const int e0 = n0 & 1023;
  const int h = e0 >> 6;             // whole 64-wide tile sits in one head
  const int half = lane >> 4, l = lane & 15;
#pragma unroll
  for (int g = 0; g < 2; ++g) {
#pragma unroll
    for (int t = 0; t < 4; ++t) {
#pragma unroll
      for (int r = 0; r < 8; ++r) {
        const int m = m0 + g * 16 + r + 8 * half;
        const int b = m >> 11, s = m & 2047;
        const int d = t * 16 + l;
        const size_t bh = (size_t)(b * HEADS + h);
        const bf16_t v = (bf16_t)(g ? c1[t][r] : c0[t][r]);
        if (sec == 0)      Qbf[(bh * SEQ + s) * HDIM + d] = v;
        else if (sec == 1) Kbf[(bh * SEQ + s) * HDIM + d] = v;
        else               Vt[(bh * HDIM + d) * SEQ + s] = v;
      }
    }
  }
}

// ---------------------------------------------------------------------------
// Flash attention.  One block = 8 waves = 8 consecutive 16-query tiles of the
// SAME (b, h).  Per 32-key step the block stages K (32x64) and V^T (64x32)
// into LDS with async b128 copies (one K + one V chunk per thread), then each
// wave runs 4 score WMMAs + online softmax + 4 output WMMAs from LDS.
// ---------------------------------------------------------------------------
__global__ __launch_bounds__(256) void attention_kernel(
    const bf16_t* __restrict__ Qbf, const bf16_t* __restrict__ Kbf,
    const bf16_t* __restrict__ Vt, const float* __restrict__ maskadd,
    bf16_t* __restrict__ attnbf) {
  __shared__ __align__(16) bf16_t Ksh[32 * HDIM];   // 4 KB: rows=key, cols=d
  __shared__ __align__(16) bf16_t Vsh[HDIM * 32];   // 4 KB: rows=d, cols=key

  const int wv = threadIdx.x >> 5;
  const int lane = threadIdx.x & 31;
  const int blk = blockIdx.x;                 // ((b*16 + h)*16 + qblk)
  const int qblk = blk & 15;
  const int h = (blk >> 4) & 15;
  const int b = blk >> 8;
  const int qt = qblk * 8 + wv;               // 16-query tile index
  const size_t bh = (size_t)(b * HEADS + h);

  const bf16_t* Qb = Qbf + bh * SEQ * HDIM;
  const bf16_t* Kb = Kbf + bh * SEQ * HDIM;
  const bf16_t* Vb = Vt + bh * HDIM * SEQ;
  const float* ma = maskadd + b * SEQ;

  // cooperative-load coordinates for this thread (one b128 chunk each)
  const int kr = threadIdx.x >> 3, kc = (threadIdx.x & 7) * 8;   // K tile
  const int vr = threadIdx.x >> 2, vc = (threadIdx.x & 3) * 8;   // V^T tile

  const v16bf qf0 = load_frag16(Qb, HDIM, qt * 16, 0);   // B-frag: Q^T d 0..31
  const v16bf qf1 = load_frag16(Qb, HDIM, qt * 16, 32);  // B-frag: Q^T d 32..63

  v8f o[4];
#pragma unroll
  for (int t = 0; t < 4; ++t) o[t] = vzero8f();
  float mrow = -3.0e38f, lrow = 0.0f;
  const int hofs = (lane & 16) ? 8 : 0;

  for (int j = 0; j < SEQ; j += 32) {
    // --- stage K/V tiles into LDS (async, whole block cooperates) ---
    async_copy_b128(&Ksh[kr * HDIM + kc], Kb + (size_t)(j + kr) * HDIM + kc);
    async_copy_b128(&Vsh[vr * 32 + vc], Vb + (size_t)vr * SEQ + j + vc);
    wait_asynccnt0();
    __syncthreads();

    // --- scores (transposed): rows = keys, cols = queries ---
    v8f s0 = vzero8f(), s1 = vzero8f();
    s0 = wmma_bf16(load_frag16(Ksh, HDIM, 0, 0), qf0, s0);
    s0 = wmma_bf16(load_frag16(Ksh, HDIM, 0, 32), qf1, s0);
    s1 = wmma_bf16(load_frag16(Ksh, HDIM, 16, 0), qf0, s1);
    s1 = wmma_bf16(load_frag16(Ksh, HDIM, 16, 32), qf1, s1);

    // mask: lane i caches maskadd[j+i]; fetch per-key value via bpermute
    const float mval = ma[j + lane];
    float f0[8], f1[8];
    float tmax = -3.0e38f;
#pragma unroll
    for (int r = 0; r < 8; ++r) {
      f0[r] = s0[r] * 0.125f + __shfl(mval, hofs + r, 32);
      f1[r] = s1[r] * 0.125f + __shfl(mval, 16 + hofs + r, 32);
      tmax = fmaxf(tmax, fmaxf(f0[r], f1[r]));
    }
    tmax = fmaxf(tmax, __shfl_xor(tmax, 16, 32));   // other 16 keys of tile
    const float mnew = fmaxf(mrow, tmax);
    const float alpha = __expf(mrow - mnew);

    float ls = 0.0f;
    v16bf pb;                                       // P^T B-fragment
#pragma unroll
    for (int r = 0; r < 8; ++r) {
      const float p0 = __expf(f0[r] - mnew);
      const float p1 = __expf(f1[r] - mnew);
      ls += p0 + p1;
      pb[r] = (bf16_t)p0;
      pb[8 + r] = (bf16_t)p1;
    }
    ls += __shfl_xor(ls, 16, 32);
    lrow = lrow * alpha + ls;
    mrow = mnew;

#pragma unroll
    for (int t = 0; t < 4; ++t) {
#pragma unroll
      for (int r = 0; r < 8; ++r) o[t][r] *= alpha;
    }
#pragma unroll
    for (int t = 0; t < 4; ++t)
      o[t] = wmma_bf16(load_frag16(Vsh, 32, t * 16, 0), pb, o[t]);

    __syncthreads();   // protect LDS tiles before next iteration overwrites
  }

  // O^T layout: lane = query, vgpr r = d0 + r (+8 for upper lanes).
  const float inv = 1.0f / lrow;
  const int q = lane & 15;
  const size_t row = (size_t)b * SEQ + qt * 16 + q;
#pragma unroll
  for (int t = 0; t < 4; ++t) {
    v8bf ov;
#pragma unroll
    for (int r = 0; r < 8; ++r) ov[r] = (bf16_t)(o[t][r] * inv);
    *(v8bf*)(attnbf + row * EMBED + h * HDIM + t * 16 + hofs) = ov;
  }
}

// ---------------------------------------------------------------------------
// Output projection: out[m, n] = sum_e attn[m, e] * w_out[n, e] + b_out[n]
// 32x64 wave tiles, fp32 output straight to d_out.
// ---------------------------------------------------------------------------
__global__ __launch_bounds__(256) void out_proj_kernel(
    const bf16_t* __restrict__ attnbf, const bf16_t* __restrict__ wbf,
    const float* __restrict__ bias, float* __restrict__ out) {
  const int wid = blockIdx.x * 8 + (threadIdx.x >> 5);
  const int lane = threadIdx.x & 31;
  const int mt = wid & 255;          // 256 M tiles of 32 rows
  const int nt = wid >> 8;           // 16 N tiles
  const int m0 = mt * 32, n0 = nt * 64;

  v8f c0[4], c1[4];
#pragma unroll
  for (int t = 0; t < 4; ++t) { c0[t] = vzero8f(); c1[t] = vzero8f(); }

  for (int k = 0; k < EMBED; k += 32) {
    v16bf a0 = load_frag16(attnbf, EMBED, m0, k);
    v16bf a1 = load_frag16(attnbf, EMBED, m0 + 16, k);
#pragma unroll
    for (int t = 0; t < 4; ++t) {
      v16bf bb = load_frag16(wbf, EMBED, n0 + 16 * t, k);
      c0[t] = wmma_bf16(a0, bb, c0[t]);
      c1[t] = wmma_bf16(a1, bb, c1[t]);
    }
  }

  const int half = lane >> 4, l = lane & 15;
#pragma unroll
  for (int t = 0; t < 4; ++t) {
    const int n = n0 + t * 16 + l;
    const float bv = bias[n];
#pragma unroll
    for (int r = 0; r < 8; ++r) {
      const int m = m0 + r + 8 * half;
      out[(size_t)m * EMBED + n] = c0[t][r] + bv;
      out[(size_t)(m + 16) * EMBED + n] = c1[t][r] + bv;
    }
  }
}

// ---------------------------------------------------------------------------
// Launch: inputs are (x, w_qkv, w_out, b_out, mask)
// ---------------------------------------------------------------------------
extern "C" void kernel_launch(void* const* d_in, const int* in_sizes, int n_in,
                              void* d_out, int out_size, void* d_ws,
                              size_t ws_size, hipStream_t stream) {
  const float* x     = (const float*)d_in[0];   // [4,2048,1024]
  const float* w_qkv = (const float*)d_in[1];   // [3072,1024]
  const float* w_out = (const float*)d_in[2];   // [1024,1024]
  const float* b_out = (const float*)d_in[3];   // [1024]
  const int*   mask  = (const int*)d_in[4];     // [4,2048]
  float* out = (float*)d_out;

  char* ws = (char*)d_ws;
  // workspace layout (bytes)
  bf16_t* xbf    = (bf16_t*)(ws + 0);                     // 16 MiB
  bf16_t* wqkvbf = (bf16_t*)(ws + (size_t)16777216);      //  6 MiB
  bf16_t* woutbf = (bf16_t*)(ws + (size_t)23068672);      //  2 MiB
  bf16_t* Qbf    = (bf16_t*)(ws + (size_t)25165824);      // 16 MiB
  bf16_t* Kbf    = (bf16_t*)(ws + (size_t)41943040);      // 16 MiB
  bf16_t* Vt     = (bf16_t*)(ws + (size_t)58720256);      // 16 MiB
  bf16_t* attnbf = (bf16_t*)(ws + (size_t)75497472);      // 16 MiB
  float*  madd   = (float*)(ws + (size_t)92274688);       // 32 KiB

  const int nx = MROWS * EMBED;          // 8388608
  const int nwq = 3 * EMBED * EMBED;     // 3145728
  const int nwo = EMBED * EMBED;         // 1048576
  const int nm = BATCH * SEQ;            // 8192

  cvt_f32_bf16_kernel<<<(nx + 255) / 256, 256, 0, stream>>>(x, xbf, nx);
  cvt_f32_bf16_kernel<<<(nwq + 255) / 256, 256, 0, stream>>>(w_qkv, wqkvbf, nwq);
  cvt_f32_bf16_kernel<<<(nwo + 255) / 256, 256, 0, stream>>>(w_out, woutbf, nwo);
  mask_bias_kernel<<<(nm + 255) / 256, 256, 0, stream>>>(mask, madd, nm);

  // QKV GEMM: 256 M-tiles x 48 N-tiles = 12288 waves -> 1536 blocks
  qkv_gemm_kernel<<<1536, 256, 0, stream>>>(xbf, wqkvbf, Qbf, Kbf, Vt);

  // Attention: 4*16*16 = 1024 blocks (8 query tiles per block, shared K/V)
  attention_kernel<<<1024, 256, 0, stream>>>(Qbf, Kbf, Vt, madd, attnbf);

  // Output projection: 256 x 16 = 4096 waves -> 512 blocks
  out_proj_kernel<<<512, 256, 0, stream>>>(attnbf, woutbf, b_out, out);
}
